// HMM_82214263980688
// MI455X (gfx1250) — matrix operand order repaired and checked by reading, not compiled, not run on hardware
//
#include <hip/hip_runtime.h>
#include <stdint.h>

// ---------------------------------------------------------------------------
// HMM forward pass for MI455X (gfx1250, wave32, WMMA).
// Dominant work: per-step [64,256]x[256,16384] GEMM -> v_wmma_f32_16x16x32_bf16
// ---------------------------------------------------------------------------

#define HN   64      // batch
#define HT   256     // sequence length
#define HV   32000   // vocab
#define HE   256     // embed dim
#define HK   128     // states
#define HTS  (HT-1)  // scan steps

typedef __bf16 v16bf __attribute__((ext_vector_type(16)));
typedef float  v8f   __attribute__((ext_vector_type(8)));

__device__ __forceinline__ unsigned short f2bf(float f) {
    union { float f; unsigned int u; } c; c.f = f;
    unsigned int u = c.u;
    u += 0x7FFFu + ((u >> 16) & 1u);      // round-to-nearest-even
    return (unsigned short)(u >> 16);
}

// ---------------- one-time preprocessing kernels ----------------

// trans_w f32 [16384,256] -> bf16 (row-major, same layout)
__global__ void hmm_cvt_trans(const float* __restrict__ w,
                              unsigned short* __restrict__ wb) {
    int idx = blockIdx.x * 256 + threadIdx.x;      // 16384*256 exact
    wb[idx] = f2bf(w[idx]);
}

// Aall[t][n][e] = bf16(embed_w[x[n][t]][e]), t in [0,255)
__global__ void hmm_gather_embed(const int* __restrict__ x,
                                 const float* __restrict__ embed_w,
                                 unsigned short* __restrict__ Aall) {
    size_t idx = (size_t)blockIdx.x * 256 + threadIdx.x;   // 255*64*256 exact
    int e = idx & 255;
    int n = (idx >> 8) & 63;
    int t = (int)(idx >> 14);
    int tok = x[n * HT + t];
    Aall[idx] = f2bf(embed_w[(size_t)tok * HE + e]);
}

// Z[kk][v] = emb_cluster_w[kk,:] . emit_w[v,:]
__global__ __launch_bounds__(256)
void hmm_emit_logits(const float* __restrict__ embc,
                     const float* __restrict__ emitw,
                     float* __restrict__ Z) {
    int v  = blockIdx.x * 256 + threadIdx.x;   // 125 blocks -> 32000 exact
    int kk = blockIdx.y;                       // 128
    __shared__ float crow[HK];
    if (threadIdx.x < HK) crow[threadIdx.x] = embc[kk * HK + threadIdx.x];
    __syncthreads();
    const float* er = emitw + (size_t)v * HK;
    float acc = 0.f;
    #pragma unroll 8
    for (int c = 0; c < HK; ++c) acc = fmaf(crow[c], er[c], acc);
    Z[(size_t)kk * HV + v] = acc;
}

// logZ[kk] = logsumexp_v Z[kk][v]
__global__ __launch_bounds__(256)
void hmm_emit_lse(const float* __restrict__ Z, float* __restrict__ logZ) {
    int kk = blockIdx.x;
    const float* row = Z + (size_t)kk * HV;
    int tid = threadIdx.x;
    __shared__ float red[8], red2[8];
    float mx = -__builtin_inff();
    for (int v = tid; v < HV; v += 256) mx = fmaxf(mx, row[v]);
    for (int off = 16; off; off >>= 1) mx = fmaxf(mx, __shfl_xor(mx, off, 32));
    if ((tid & 31) == 0) red[tid >> 5] = mx;
    __syncthreads();
    float m2 = red[0];
    #pragma unroll
    for (int w = 1; w < 8; ++w) m2 = fmaxf(m2, red[w]);
    float s = 0.f;
    for (int v = tid; v < HV; v += 256) s += __expf(row[v] - m2);
    for (int off = 16; off; off >>= 1) s += __shfl_xor(s, off, 32);
    if ((tid & 31) == 0) red2[tid >> 5] = s;
    __syncthreads();
    if (tid == 0) {
        float tot = 0.f;
        #pragma unroll
        for (int w = 0; w < 8; ++w) tot += red2[w];
        logZ[kk] = m2 + __logf(tot);
    }
}

// E2[t][n][j] = Z[j][x[n][t+1]] - logZ[j]   (log emission of observed token)
__global__ void hmm_gather_emit(const int* __restrict__ x,
                                const float* __restrict__ Z,
                                const float* __restrict__ logZ,
                                float* __restrict__ E2) {
    size_t idx = (size_t)blockIdx.x * 256 + threadIdx.x;   // 255*64*128 exact
    int j = idx & 127;
    int n = (idx >> 7) & 63;
    int t = (int)(idx >> 13);
    int tok = x[n * HT + (t + 1)];
    E2[idx] = Z[(size_t)j * HV + tok] - logZ[j];
}

// pre[n][j] = log_softmax(start_w[:,0]+start_b)[j]; m[n]=max_j; acc=0
__global__ __launch_bounds__(128)
void hmm_init_pre(const float* __restrict__ start_w,
                  const float* __restrict__ start_b,
                  float* __restrict__ pre, float* __restrict__ mrow,
                  float* __restrict__ acc) {
    int j = threadIdx.x;
    float s = start_w[j] + start_b[j];
    __shared__ float red[4], red2[4];
    float mx = s;
    for (int off = 16; off; off >>= 1) mx = fmaxf(mx, __shfl_xor(mx, off, 32));
    if ((j & 31) == 0) red[j >> 5] = mx;
    __syncthreads();
    mx = fmaxf(fmaxf(red[0], red[1]), fmaxf(red[2], red[3]));
    float e = __expf(s - mx);
    float sum = e;
    for (int off = 16; off; off >>= 1) sum += __shfl_xor(sum, off, 32);
    if ((j & 31) == 0) red2[j >> 5] = sum;
    __syncthreads();
    sum = red2[0] + red2[1] + red2[2] + red2[3];
    float lse = mx + __logf(sum);
    float p0 = s - lse;
    for (int n = 0; n < HN; ++n) { pre[n * HK + j] = p0; acc[n * HK + j] = 0.f; }
    if (j < HN) mrow[j] = mx - lse;   // max_j p0
}

// ---------------- per-step kernels ----------------

// One workgroup per source-state block i (128 blocks, 256 threads = 8 waves).
// S[n, j] = sum_e A[n,e] * W[i*128+j, e]   via WMMA bf16, f32 accumulate.
// Then tran[n,j] = S - lse_j(S), and atomic acc[n,j] += exp(pre[n,i]-m[n]+tran).
__global__ __launch_bounds__(256)
void hmm_step_gemm(const unsigned short* __restrict__ Aall,  // [255][64][256] bf16
                   const unsigned short* __restrict__ Wb,    // [16384][256]  bf16
                   const float* __restrict__ pre,            // [64][128]
                   const float* __restrict__ mrow,           // [64]
                   float* __restrict__ acc,                  // [64][128]
                   int t) {
    const int i    = blockIdx.x;       // source state block 0..127
    const int tid  = threadIdx.x;
    const int wave = tid >> 5;
    const int lane = tid & 31;

    // LDS: A 64x264 bf16 (33,792B) | B 128x264 bf16 (67,584B)  => 101,376B
    // After the GEMM, S (64x132 f32, 33,792B) reuses the B region.
    extern __shared__ unsigned char smem[];
    uint32_t* Al  = (uint32_t*)smem;                  // row stride 132 dwords
    uint32_t* Bl  = (uint32_t*)(smem + 64 * 264 * 2); // row stride 132 dwords
    float*    Ssm = (float*)   (smem + 64 * 264 * 2); // row stride 132 floats

    // --- stage A (this step's 64x256 bf16) and B (W rows i*128..i*128+127) ---
    const uint4* Ag = (const uint4*)(Aall + (size_t)t * HN * HE);   // 2048 uint4
    #pragma unroll
    for (int r = 0; r < 8; ++r) {
        int gi  = (r << 8) + tid;
        int row = gi >> 5, c4 = gi & 31;                 // 32 uint4 per row
        *(uint4*)(Al + row * 132 + (c4 << 2)) = Ag[gi];
    }
    const uint4* Bg = (const uint4*)(Wb + (size_t)i * HK * HE);     // 4096 uint4
    #pragma unroll
    for (int r = 0; r < 16; ++r) {
        int gi  = (r << 8) + tid;
        int row = gi >> 5, c4 = gi & 31;
        *(uint4*)(Bl + row * 132 + (c4 << 2)) = Bg[gi];
    }
    __syncthreads();

    // --- WMMA: wave owns M-tile mt (16 rows of n) x 4 N-tiles (64 cols of j) ---
    const int mt = wave >> 1;
    const int nh = wave & 1;

    v8f Cacc[4] = {};
    const int arow = mt * 16 + (lane & 15);
    const int alo  = (lane & 16) ? 8 : 0;
    const uint32_t* Ar = Al + arow * 132;
    const int bko_lane = (lane & 16) ? 8 : 0;

    #pragma unroll
    for (int ks = 0; ks < 8; ++ks) {
        const int k0 = ks * 32;
        // A fragment: 16x32 bf16 per ISA layout (lane<16: K=0..7,16..23; lane>=16: +8)
        v16bf afrag;
        uint32_t* au = (uint32_t*)&afrag;
        #pragma unroll
        for (int h = 0; h < 8; ++h) {
            int k = ((h < 4) ? (2 * h) : (16 + 2 * (h - 4))) + alo + k0;
            au[h] = Ar[k >> 1];
        }
        #pragma unroll
        for (int q = 0; q < 4; ++q) {
            const int j0 = (nh * 4 + q) * 16;
            // B fragment: 32x16; col = lane&15, K = 0..7,16..23 (lanes>=16: +8), same as A
            v16bf bfrag;
            uint32_t* bu = (uint32_t*)&bfrag;
            const uint32_t* Br = Bl + (j0 + (lane & 15)) * 132;
            #pragma unroll
            for (int h = 0; h < 8; ++h) {
                int k = ((h < 4) ? (2 * h) : (16 + 2 * (h - 4))) + bko_lane + k0;
                bu[h] = Br[k >> 1];
            }
            Cacc[q] = __builtin_amdgcn_wmma_f32_16x16x32_bf16(
                false, afrag, false, bfrag, (short)0, Cacc[q], false, false);
        }
    }
    __syncthreads();   // all B reads done; reuse region for S

    // --- scatter C tiles to Ssm (C layout: col=lane&15, row=vgpr + (lane>=16?8:0)) ---
    #pragma unroll
    for (int q = 0; q < 4; ++q) {
        const int j0    = (nh * 4 + q) * 16 + (lane & 15);
        const int rbase = mt * 16 + ((lane & 16) ? 8 : 0);
        const float* cf = (const float*)&Cacc[q];
        #pragma unroll
        for (int r = 0; r < 8; ++r)
            Ssm[(rbase + r) * 132 + j0] = cf[r];
    }
    __syncthreads();

    // --- per-row log-softmax over j (local), then atomic accumulate exp() ---
    // wave w handles rows n = 8w .. 8w+7
    for (int rr = 0; rr < 8; ++rr) {
        const int n = wave * 8 + rr;
        const float* Srow = &Ssm[n * 132];
        float v0 = Srow[lane], v1 = Srow[lane + 32],
              v2 = Srow[lane + 64], v3 = Srow[lane + 96];
        float mx = fmaxf(fmaxf(v0, v1), fmaxf(v2, v3));
        for (int off = 16; off; off >>= 1) mx = fmaxf(mx, __shfl_xor(mx, off, 32));
        float s = __expf(v0 - mx) + __expf(v1 - mx) + __expf(v2 - mx) + __expf(v3 - mx);
        for (int off = 16; off; off >>= 1) s += __shfl_xor(s, off, 32);
        float lse = mx + __logf(s);
        // exp(pre[n,i] - m[n] + (S - lse)); m[n] keeps the exp in range
        float scale = pre[n * HK + i] - mrow[n] - lse;
        atomicAdd(&acc[n * HK + lane],      __expf(scale + v0));
        atomicAdd(&acc[n * HK + lane + 32], __expf(scale + v1));
        atomicAdd(&acc[n * HK + lane + 64], __expf(scale + v2));
        atomicAdd(&acc[n * HK + lane + 96], __expf(scale + v3));
    }
}

// pre[n][j] = m[n] + log(acc[n][j]) + E2[t][n][j]; m[n] = max_j; acc = 0
__global__ __launch_bounds__(128)
void hmm_step_finalize(float* __restrict__ pre, float* __restrict__ mrow,
                       float* __restrict__ acc, const float* __restrict__ E2,
                       int t) {
    const int n = blockIdx.x;
    const int j = threadIdx.x;
    float mOld = mrow[n];
    float a = acc[n * HK + j];
    float v = mOld + __logf(a) + E2[((size_t)t * HN + n) * HK + j];
    pre[n * HK + j] = v;
    acc[n * HK + j] = 0.f;
    __shared__ float red[4];
    float mx = v;
    for (int off = 16; off; off >>= 1) mx = fmaxf(mx, __shfl_xor(mx, off, 32));
    if ((j & 31) == 0) red[j >> 5] = mx;
    __syncthreads();
    if (j == 0)
        mrow[n] = fmaxf(fmaxf(red[0], red[1]), fmaxf(red[2], red[3]));
}

// out = -mean_n logsumexp_j pre[n][j]
__global__ __launch_bounds__(128)
void hmm_loss(const float* __restrict__ pre, float* __restrict__ out) {
    const int j = threadIdx.x;
    __shared__ float red[4], red2[4];
    float total = 0.f;
    for (int n = 0; n < HN; ++n) {
        float v = pre[n * HK + j];
        float mx = v;
        for (int off = 16; off; off >>= 1) mx = fmaxf(mx, __shfl_xor(mx, off, 32));
        if ((j & 31) == 0) red[j >> 5] = mx;
        __syncthreads();
        mx = fmaxf(fmaxf(red[0], red[1]), fmaxf(red[2], red[3]));
        float s = __expf(v - mx);
        for (int off = 16; off; off >>= 1) s += __shfl_xor(s, off, 32);
        if ((j & 31) == 0) red2[j >> 5] = s;
        __syncthreads();
        total += mx + __logf(red2[0] + red2[1] + red2[2] + red2[3]);
        __syncthreads();
    }
    if (j == 0) out[0] = -total / (float)HN;
}

// ---------------------------------------------------------------------------

extern "C" void kernel_launch(void* const* d_in, const int* in_sizes, int n_in,
                              void* d_out, int out_size, void* d_ws, size_t ws_size,
                              hipStream_t stream) {
    (void)in_sizes; (void)n_in; (void)out_size; (void)ws_size;
    const int*   x        = (const int*)  d_in[0];   // [64,256]
    const float* embed_w  = (const float*)d_in[1];   // [32000,256]
    const float* trans_w  = (const float*)d_in[2];   // [16384,256]
    const float* start_w  = (const float*)d_in[3];   // [128,1]
    const float* start_b  = (const float*)d_in[4];   // [128]
    const float* embc     = (const float*)d_in[5];   // [128,128]
    const float* emitw    = (const float*)d_in[6];   // [32000,128]
    float*       out      = (float*)d_out;

    // workspace carve (256B aligned), total ~41.5 MB
    char* ws = (char*)d_ws;
    size_t o = 0;
    auto carve = [&](size_t bytes) -> char* {
        char* p = ws + o;
        o = (o + bytes + 255) & ~(size_t)255;
        return p;
    };
    unsigned short* Wb   = (unsigned short*)carve((size_t)16384 * 256 * 2);
    unsigned short* Aall = (unsigned short*)carve((size_t)HTS * HN * HE * 2);
    float* E2   = (float*)carve((size_t)HTS * HN * HK * 4);
    float* Z    = (float*)carve((size_t)HK * HV * 4);
    float* logZ = (float*)carve((size_t)HK * 4);
    float* pre  = (float*)carve((size_t)HN * HK * 4);
    float* mrow = (float*)carve((size_t)HN * 4);
    float* acc  = (float*)carve((size_t)HN * HK * 4);

    // one-time preprocessing
    hmm_cvt_trans   <<<16384, 256, 0, stream>>>(trans_w, Wb);
    hmm_gather_embed<<<(HTS * HN * HE) / 256, 256, 0, stream>>>(x, embed_w, Aall);
    hmm_emit_logits <<<dim3(HV / 256, HK), 256, 0, stream>>>(embc, emitw, Z);
    hmm_emit_lse    <<<HK, 256, 0, stream>>>(Z, logZ);
    hmm_gather_emit <<<(HTS * HN * HK) / 256, 256, 0, stream>>>(x, Z, logZ, E2);
    hmm_init_pre    <<<1, 128, 0, stream>>>(start_w, start_b, pre, mrow, acc);

    // sequential forward scan: WMMA GEMM + combine per step
    const size_t shmem = (size_t)(64 * 264 + 128 * 264) * 2;   // 101,376 B
    for (int t = 0; t < HTS; ++t) {
        hmm_step_gemm    <<<HK, 256, shmem, stream>>>(Aall, Wb, pre, mrow, acc, t);
        hmm_step_finalize<<<HN, 128, 0, stream>>>(pre, mrow, acc, E2, t);
    }
    hmm_loss<<<1, 128, 0, stream>>>(pre, out);
}
